// VQ_46600395162277
// MI455X (gfx1250) — matrix operand order repaired and checked by reading, not compiled, not run on hardware
//
#include <hip/hip_runtime.h>

typedef __attribute__((ext_vector_type(16))) __bf16 v16bf;
typedef __attribute__((ext_vector_type(8)))  float  v8f;
typedef __attribute__((ext_vector_type(4)))  int    v4i;

#define K_CB   4096
#define D_DIM  256
#define BT_N   16384
#define BLK_M  128
#define NTILE  64
#define NCHUNK (K_CB / NTILE)
#define BETA_F 0.2f

#if defined(__has_builtin)
#if __has_builtin(__builtin_amdgcn_global_load_async_to_lds_b128) && \
    __has_builtin(__builtin_amdgcn_s_wait_asynccnt)
#define USE_ASYNC_LDS 1
#endif
#endif
#ifndef USE_ASYNC_LDS
#define USE_ASYNC_LDS 0
#endif

#if USE_ASYNC_LDS
#define ASYNC_FENCE() __builtin_amdgcn_s_wait_asynccnt(0)
// builtin expects ext-vector int4 pointers: global (AS1) src, LDS (AS3) dst
typedef __attribute__((address_space(1))) v4i gv4i;
typedef __attribute__((address_space(3))) v4i lv4i;
#else
#define ASYNC_FENCE()
#endif

__device__ __forceinline__ unsigned short f2bf_bits(float f) {
    unsigned u = __builtin_bit_cast(unsigned, f);
    unsigned r = (u + 0x7FFFu + ((u >> 16) & 1u)) >> 16;   // RNE
    return (unsigned short)r;
}
__device__ __forceinline__ __bf16 f2bf(float f) {
    unsigned short h = f2bf_bits(f);
    return __builtin_bit_cast(__bf16, h);
}

// ---------------------------------------------------------------------------
// Kernel 1: embedding f32 -> bf16 (workspace) + halfnorm[k] = 0.5*||e_k||^2
// ---------------------------------------------------------------------------
__global__ void __launch_bounds__(256) vq_prep(const float* __restrict__ emb,
                                               __bf16* __restrict__ ebf,
                                               float* __restrict__ hn) {
    const int wave = threadIdx.x >> 5;
    const int lane = threadIdx.x & 31;
    const int row  = blockIdx.x * 8 + wave;

    const float* src = emb + (size_t)row * D_DIM + lane * 8;
    float4 v0 = ((const float4*)src)[0];
    float4 v1 = ((const float4*)src)[1];
    float vals[8] = {v0.x, v0.y, v0.z, v0.w, v1.x, v1.y, v1.z, v1.w};

    float s = 0.f;
    unsigned short h[8];
#pragma unroll
    for (int i = 0; i < 8; ++i) { s += vals[i] * vals[i]; h[i] = f2bf_bits(vals[i]); }

    uint4 packed;
    packed.x = (unsigned)h[0] | ((unsigned)h[1] << 16);
    packed.y = (unsigned)h[2] | ((unsigned)h[3] << 16);
    packed.z = (unsigned)h[4] | ((unsigned)h[5] << 16);
    packed.w = (unsigned)h[6] | ((unsigned)h[7] << 16);
    *(uint4*)(ebf + (size_t)row * D_DIM + lane * 8) = packed;

#pragma unroll
    for (int off = 16; off >= 1; off >>= 1) s += __shfl_xor(s, off, 32);
    if (lane == 0) hn[row] = 0.5f * s;
}

// ---------------------------------------------------------------------------
// Kernel 2: fused bf16-WMMA GEMM + running argmin + gather + loss partial
// 256 threads = 8 waves; wave w owns rows [blk*128 + 16w, +16).
// Double-buffered LDS codeword tiles streamed via the async-to-LDS pipe.
// ---------------------------------------------------------------------------
__global__ void __launch_bounds__(256) vq_main(
    const float* __restrict__ x, const float* __restrict__ emb,
    const __bf16* __restrict__ ebf, const float* __restrict__ hn,
    float* __restrict__ out, float* __restrict__ idx_out,
    float* __restrict__ partial) {

    // 2 x 32KB ping-pong codeword tiles == exactly 64KB static LDS
    __shared__ __align__(32) __bf16 Blds[2][NTILE * D_DIM];

    const int tid  = threadIdx.x;
    const int wave = tid >> 5;
    const int lane = tid & 31;
    const int l15  = lane & 15;
    const int hi   = lane >> 4;           // 0: low half-wave, 1: high half-wave
    const int row0 = blockIdx.x * BLK_M;

    // --- A fragments: 16 rows x 256 k of x, bf16, kept in VGPRs all kernel ---
    // 16-bit A 16x32 layout (matches B): M = lane%16; lanes<16 hold K 0..15,
    // lanes>=16 hold K 16..31 (per 32-k step s).
    v16bf afrag[8];
    {
        const float* xrow = x + (size_t)(row0 + wave * 16 + l15) * D_DIM;
        const int khalf = hi * 16;
#pragma unroll
        for (int s = 0; s < 8; ++s) {
#pragma unroll
            for (int j = 0; j < 8; ++j)
                afrag[s][j]     = f2bf(xrow[32 * s + khalf + j]);
#pragma unroll
            for (int j = 0; j < 8; ++j)
                afrag[s][8 + j] = f2bf(xrow[32 * s + khalf + 8 + j]);
        }
    }

    // stage one 64-codeword bf16 tile (32KB) into LDS buffer `buf`
    auto stage_tile = [&](int c, int buf) {
        const uint4* srcg = (const uint4*)(ebf + (size_t)c * NTILE * D_DIM) + tid;
#if USE_ASYNC_LDS
        gv4i* gp = (gv4i*)srcg;
        lv4i* lp = (lv4i*)((uint4*)&Blds[buf][0] + tid);
#pragma unroll
        for (int i = 0; i < 8; ++i)
            __builtin_amdgcn_global_load_async_to_lds_b128(gp + 256 * i,
                                                           lp + 256 * i, 0, 0);
#else
        uint4* dst = (uint4*)&Blds[buf][0] + tid;
#pragma unroll
        for (int i = 0; i < 8; ++i) dst[256 * i] = srcg[256 * i];
#endif
    };

    float minval[8];
    int   minidx[8];
#pragma unroll
    for (int r = 0; r < 8; ++r) { minval[r] = 3.4e38f; minidx[r] = 0; }

    stage_tile(0, 0);
    ASYNC_FENCE();
    __syncthreads();

    for (int c = 0; c < NCHUNK; ++c) {
        const int buf = c & 1;
        if (c + 1 < NCHUNK) stage_tile(c + 1, buf ^ 1);

        // two independent accumulator chains per pass -> WMMA pipe stays busy
#pragma unroll
        for (int t = 0; t < NTILE / 16; t += 2) {
            v8f acc0 = {};
            v8f acc1 = {};
            // 16-bit B 32x16 layout used here (same k order as A): N = lane%16; lanes<16 hold K 0..15,
            // lanes>=16 hold K 16..31 -> one contiguous 32B LDS read per frag.
            const __bf16* b0 = &Blds[buf][0] + (t * 16 + l15) * D_DIM + hi * 16;
            const __bf16* b1 = b0 + 16 * D_DIM;
#pragma unroll
            for (int s = 0; s < 8; ++s) {
                v16bf vb0 = *(const v16bf*)(b0 + 32 * s);
                v16bf vb1 = *(const v16bf*)(b1 + 32 * s);
                acc0 = __builtin_amdgcn_wmma_f32_16x16x32_bf16(
                         false, afrag[s], false, vb0, (short)0, acc0, false, false);
                acc1 = __builtin_amdgcn_wmma_f32_16x16x32_bf16(
                         false, afrag[s], false, vb1, (short)0, acc1, false, false);
            }
            const int   n0 = c * NTILE + t * 16 + l15;
            const int   n1 = n0 + 16;
            const float h0 = hn[n0];                  // L2-resident broadcast row
            const float h1 = hn[n1];
#pragma unroll
            for (int r = 0; r < 8; ++r) {
                float sc0 = h0 - acc0[r];             // 0.5*dist - 0.5*||x||^2
                if (sc0 < minval[r]) { minval[r] = sc0; minidx[r] = n0; }
                float sc1 = h1 - acc1[r];
                if (sc1 < minval[r]) { minval[r] = sc1; minidx[r] = n1; }
            }
        }

        ASYNC_FENCE();          // next buffer fully resident in LDS
        __syncthreads();        // and everyone done reading this buffer
    }

    // GEMM done: reuse Blds as index / reduction scratch (barrier above covers)
    int*   idxLds = (int*)&Blds[0][0];
    float* redLds = (float*)&Blds[0][0] + BLK_M;

    // reduce (min, argmin) over the 16 lanes sharing each row; halves stay
    // independent (xor masks < 16): rows m0+r (lane 0) and m0+8+r (lane 16)
#pragma unroll
    for (int r = 0; r < 8; ++r) {
#pragma unroll
        for (int off = 8; off >= 1; off >>= 1) {
            float ov = __shfl_xor(minval[r], off, 32);
            int   oi = __shfl_xor(minidx[r], off, 32);
            if (ov < minval[r] || (ov == minval[r] && oi < minidx[r])) {
                minval[r] = ov; minidx[r] = oi;        // first-index tie-break
            }
        }
        if (l15 == 0) idxLds[wave * 16 + hi * 8 + r] = minidx[r];
    }
    __syncthreads();

    // gather winning codewords in exact fp32, write out + indexes, loss partial
    float lsum = 0.f;
    for (int r = 0; r < BLK_M; ++r) {
        const int    gi = idxLds[r];
        const float  ev = emb[(size_t)gi * D_DIM + tid];       // tid == d
        const size_t go = (size_t)(row0 + r) * D_DIM + tid;
        const float  xv = x[go];
        out[go] = ev;                                          // STE forward
        const float d = xv - ev;
        lsum += d * d;
    }
    if (tid < BLK_M) idx_out[row0 + tid] = (float)idxLds[tid];

#pragma unroll
    for (int off = 16; off >= 1; off >>= 1) lsum += __shfl_xor(lsum, off, 32);
    if (lane == 0) redLds[wave] = lsum;
    __syncthreads();
    if (tid == 0) {
        float s = 0.f;
#pragma unroll
        for (int i = 0; i < 8; ++i) s += redLds[i];
        partial[blockIdx.x] = s;
    }
}

// ---------------------------------------------------------------------------
// Kernel 3: reduce 128 partials -> loss = (1+beta) * mean((x - values)^2)
// ---------------------------------------------------------------------------
__global__ void __launch_bounds__(128) vq_loss(const float* __restrict__ partial,
                                               float* __restrict__ loss_out) {
    __shared__ float red[4];
    const int tid = threadIdx.x;
    float s = partial[tid];
#pragma unroll
    for (int off = 16; off >= 1; off >>= 1) s += __shfl_xor(s, off, 32);
    if ((tid & 31) == 0) red[tid >> 5] = s;
    __syncthreads();
    if (tid == 0) {
        float t = red[0] + red[1] + red[2] + red[3];
        loss_out[0] = (1.0f + BETA_F) * t / (float)((size_t)BT_N * D_DIM);
    }
}

extern "C" void kernel_launch(void* const* d_in, const int* in_sizes, int n_in,
                              void* d_out, int out_size, void* d_ws, size_t ws_size,
                              hipStream_t stream) {
    const float* x   = (const float*)d_in[0];   // [8,2048,256] f32
    const float* emb = (const float*)d_in[1];   // [4096,256]   f32

    char* ws = (char*)d_ws;
    __bf16* ebf    = (__bf16*)ws;                                    // 2 MB
    float*  hn     = (float*)(ws + (size_t)K_CB * D_DIM * 2);        // 16 KB
    float*  parts  = (float*)(ws + (size_t)K_CB * D_DIM * 2 + K_CB * 4);

    float* out      = (float*)d_out;                 // [BT, D]
    float* idx_out  = out + (size_t)BT_N * D_DIM;    // [BT] (as f32)
    float* loss_out = idx_out + BT_N;                // [1]

    vq_prep<<<K_CB / 8, 256, 0, stream>>>(emb, ebf, hn);
    vq_main<<<BT_N / BLK_M, 256, 0, stream>>>(x, emb, ebf, hn, out, idx_out, parts);
    vq_loss<<<1, 128, 0, stream>>>(parts, loss_out);
}
